// Edge_cycle_50869592655502
// MI455X (gfx1250) — matrix operand order repaired and checked by reading, not compiled, not run on hardware
//
#include <hip/hip_runtime.h>
#include <hip/hip_bf16.h>

// ---------------------------------------------------------------------------
// MI455X (gfx1250) implementation of the Autobahn edge/cycle GNN layer.
// All dense matmuls run on v_wmma_f32_16x16x32_bf16 (bf16 A/B, f32 accum):
// ~117 GFLOP total vs ~1.5 GB streamed -> memory-bound at 23.3 TB/s. The
// GEMM tile covers the full output width per wave (A read exactly once) and
// all B fragments of a k-step are loaded before the WMMA group so loadcnt
// waits drain progressively instead of serializing load->wmma per tile.
// ---------------------------------------------------------------------------

typedef __bf16 bh;
typedef __attribute__((ext_vector_type(16))) __bf16 v16bf;
typedef __attribute__((ext_vector_type(8)))  __bf16 v8bf;
typedef __attribute__((ext_vector_type(8)))  float  v8f;

#define NSZ   6
#define MCYC  10000
#define NEDGE 400000
#define RCY   330000      /* total cycle-node rows = M * (3+4+5+6+7+8) */
#define DD    64
#define NCH   4
#define HH    128
#define CIN   320         /* (NCH+1)*DD */
#define NBLK  240         /* column-stat partial blocks (deterministic) */

// ---------------- fragment helpers (documented CDNA5 WMMA layouts) ----------

// Build a v16bf fragment from two contiguous 8-element (16B) chunks.
__device__ __forceinline__ v16bf make_frag(const bh* lo, const bh* hi) {
  v8bf a = *(const v8bf*)lo;
  v8bf b = *(const v8bf*)hi;
  v16bf r;
#pragma unroll
  for (int e = 0; e < 8; ++e) { r[e] = a[e]; r[e + 8] = b[e]; }
  return r;
}

// ---------------- generic 16x(NT*16)-per-wave bf16 WMMA GEMM ----------------
// C[row0:row0+16, 0:NT*16] = A[rows,K] @ B[K,N], f32 out. K%32==0, N==NT*16.
template <int NT>
__global__ void __launch_bounds__(32)
k_gemm_bf16(const bh* __restrict__ A, const bh* __restrict__ B,
            float* __restrict__ C, int K, int N) {
  const int lane  = threadIdx.x & 31;
  const int l15   = lane & 15;
  const int hioff = (lane >> 4) << 3;            // +8 for lanes 16..31
  const int row0  = blockIdx.x * 16;
  const bh* arow  = A + (size_t)(row0 + l15) * K;
  v8f acc[NT];
#pragma unroll
  for (int t = 0; t < NT; ++t) acc[t] = v8f{};
  for (int k0 = 0; k0 < K; k0 += 32) {
    if (k0 + 32 < K) __builtin_prefetch(arow + k0 + 32 + hioff, 0, 1);
    // cluster ALL loads of this k-step before the WMMA group
    v16bf a = make_frag(arow + k0 + hioff, arow + k0 + hioff + 16);
    const bh* bk = B + (size_t)(k0 + hioff) * N + l15; // B: n = l15, table-order k
    v16bf bb[NT];
#pragma unroll
    for (int t = 0; t < NT; ++t)
#pragma unroll
      for (int e = 0; e < 8; ++e) {
        bb[t][e] = bk[(size_t)e * N + t * 16];
        bb[t][e + 8] = bk[(size_t)(16 + e) * N + t * 16];
      }
#pragma unroll
    for (int t = 0; t < NT; ++t)
      acc[t] = __builtin_amdgcn_wmma_f32_16x16x32_bf16(
          false, a, false, bb[t], (short)0, acc[t], false, false);
  }
  float* crow = C + (size_t)(row0 + hioff) * N + l15;
#pragma unroll
  for (int t = 0; t < NT; ++t)
#pragma unroll
    for (int v = 0; v < 8; ++v)
      crow[(size_t)v * N + t * 16] = acc[t][v];
}

// ---------------- Autobahn cyclic conv (per channel, per cycle size) --------
// y[m,j,:] = sum_sh nodes[m,(j+sh)%k,:] @ W[c,s,sh]   (k matmuls of 64x64)
// Output written as bf16 into cyc_in columns [64 + c*64, 64 + c*64 + 64).
__global__ void __launch_bounds__(32)
k_autobahn(const bh* __restrict__ nodes, const bh* __restrict__ W,
           bh* __restrict__ cycin, int seg_base, int k, int s_idx) {
  const int lane  = threadIdx.x & 31;
  const int l15   = lane & 15;
  const int hioff = (lane >> 4) << 3;
  const int tile  = blockIdx.x;
  const int c     = blockIdx.y;
  const int rloc  = tile * 16 + l15;
  const int m     = rloc / k;
  const int j     = rloc - m * k;
  const bh* seg = nodes + (size_t)seg_base * DD;
  const bh* Wc  = W + ((size_t)c * NSZ + s_idx) * 8 * (DD * DD);
  v8f acc[4] = {};
  for (int sh = 0; sh < k; ++sh) {
    int jr = j + sh; if (jr >= k) jr -= k;               // cyclic roll
    const bh* arow = seg + ((size_t)m * k + jr) * DD;
    const bh* Ws   = Wc + (size_t)sh * (DD * DD);
    // cluster all loads of this shift (2 A frags + 8 B frags), then 8 WMMAs
    v16bf a0 = make_frag(arow + hioff, arow + hioff + 16);
    v16bf a1 = make_frag(arow + 32 + hioff, arow + 32 + hioff + 16);
    const bh* bk0 = Ws + (size_t)hioff * DD + l15;
    const bh* bk1 = Ws + (size_t)(32 + hioff) * DD + l15;
    v16bf b0[4], b1[4];
#pragma unroll
    for (int t = 0; t < 4; ++t) {
#pragma unroll
      for (int e = 0; e < 8; ++e) {
        b0[t][e] = bk0[(size_t)e * DD + t * 16];
        b0[t][e + 8] = bk0[(size_t)(16 + e) * DD + t * 16];
        b1[t][e] = bk1[(size_t)e * DD + t * 16];
        b1[t][e + 8] = bk1[(size_t)(16 + e) * DD + t * 16];
      }
    }
#pragma unroll
    for (int t = 0; t < 4; ++t)
      acc[t] = __builtin_amdgcn_wmma_f32_16x16x32_bf16(
          false, a0, false, b0[t], (short)0, acc[t], false, false);
#pragma unroll
    for (int t = 0; t < 4; ++t)
      acc[t] = __builtin_amdgcn_wmma_f32_16x16x32_bf16(
          false, a1, false, b1[t], (short)0, acc[t], false, false);
  }
  bh* crow = cycin + ((size_t)seg_base + tile * 16 + hioff) * CIN
                   + (DD + c * DD) + l15;
#pragma unroll
  for (int t = 0; t < 4; ++t)
#pragma unroll
    for (int v = 0; v < 8; ++v)
      crow[(size_t)v * CIN + t * 16] = (bh)acc[t][v];
}

// ---------------- elementwise / gather / scatter / BN -----------------------

__global__ void k_f32_to_bf16(const float* __restrict__ X, bh* __restrict__ Y,
                              size_t n) {
  size_t i = (size_t)blockIdx.x * blockDim.x + threadIdx.x;
  if (i < n) Y[i] = (bh)X[i];
}

// copy f32 [rows, sc] into bf16 [rows, ldy] at column offset col0 (sc pow2)
__global__ void k_cast_cols_bf16(const float* __restrict__ X, bh* __restrict__ Y,
                                 size_t n, int sc, int ldy, int col0) {
  size_t i = (size_t)blockIdx.x * blockDim.x + threadIdx.x;
  if (i >= n) return;
  int col = (int)(i & (size_t)(sc - 1));
  size_t row = i / (size_t)sc;
  Y[row * ldy + col0 + col] = (bh)X[i];
}

__global__ void k_zero_f32(float* __restrict__ p, size_t n) {
  size_t i = (size_t)blockIdx.x * blockDim.x + threadIdx.x;
  if (i < n) p[i] = 0.0f;
}

// node[m,j] = edge_rep[ce[m,j]] + edge_rep[ce[m,j-1 mod k]]  -> bf16
__global__ void k_gather_nodes(const float* __restrict__ er,
                               const int* __restrict__ ce, bh* __restrict__ nodes,
                               int seg_base, int k, int rows_local) {
  int tid = blockIdx.x * blockDim.x + threadIdx.x;
  int col = tid & 63;
  int r   = tid >> 6;
  if (r >= rows_local) return;
  int m = r / k, j = r - m * k;
  int jm = j ? j - 1 : k - 1;
  int e0 = ce[m * k + j], e1 = ce[m * k + jm];
  float v = er[(size_t)e0 * DD + col] + er[(size_t)e1 * DD + col];
  nodes[((size_t)seg_base + r) * DD + col] = (bh)v;
}

// cycle2edge[ce[m,j]] += cycle_out[m,j] + cycle_out[m,(j+1)%k]
__global__ void k_scatter(const float* __restrict__ co, const int* __restrict__ ce,
                          float* __restrict__ c2e, int seg_base, int k,
                          int rows_local) {
  int tid = blockIdx.x * blockDim.x + threadIdx.x;
  int col = tid & 63;
  int r   = tid >> 6;
  if (r >= rows_local) return;
  int m = r / k, j = r - m * k;
  int jn = (j == k - 1) ? 0 : j + 1;
  float v = co[((size_t)seg_base + r) * DD + col]
          + co[((size_t)seg_base + m * k + jn) * DD + col];
  atomicAdd(c2e + (size_t)ce[m * k + j] * DD + col, v);
}

// deterministic column stats: per-block partial sum/sumsq (blockDim == ncols)
__global__ void k_colstat_part(const float* __restrict__ X,
                               float* __restrict__ part, int rows, int ncols) {
  int col = threadIdx.x;
  int per = (rows + gridDim.x - 1) / gridDim.x;
  int r0 = blockIdx.x * per;
  int r1 = r0 + per; if (r1 > rows) r1 = rows;
  float s = 0.f, q = 0.f;
  for (int r = r0; r < r1; ++r) {
    float x = X[(size_t)r * ncols + col];
    s += x; q += x * x;
  }
  part[(size_t)blockIdx.x * 2 * ncols + col] = s;
  part[(size_t)blockIdx.x * 2 * ncols + ncols + col] = q;
}

__global__ void k_colstat_fin(const float* __restrict__ part,
                              const float* __restrict__ g,
                              const float* __restrict__ b,
                              float* __restrict__ scale, float* __restrict__ shift,
                              int rows, int nb, int ncols) {
  int col = threadIdx.x;
  float s = 0.f, q = 0.f;
  for (int i = 0; i < nb; ++i) {
    s += part[(size_t)i * 2 * ncols + col];
    q += part[(size_t)i * 2 * ncols + ncols + col];
  }
  float inv = 1.0f / (float)rows;
  float mean = s * inv;
  float var  = q * inv - mean * mean;         // biased var, matches jnp.var
  float sc = g[col] * rsqrtf(var + 1e-5f);
  scale[col] = sc;
  shift[col] = b[col] - mean * sc;
}

__global__ void k_bn_relu_bf16(const float* __restrict__ X,
                               const float* __restrict__ sc,
                               const float* __restrict__ sh,
                               bh* __restrict__ Y, size_t n, int ncols) {
  size_t i = (size_t)blockIdx.x * blockDim.x + threadIdx.x;
  if (i >= n) return;
  int col = (int)(i & (size_t)(ncols - 1));
  float v = fmaf(X[i], sc[col], sh[col]);
  Y[i] = (bh)(v > 0.f ? v : 0.f);
}

__global__ void k_bn_relu_f32(const float* __restrict__ X,
                              const float* __restrict__ sc,
                              const float* __restrict__ sh,
                              float* __restrict__ Y, size_t n, int ncols) {
  size_t i = (size_t)blockIdx.x * blockDim.x + threadIdx.x;
  if (i >= n) return;
  int col = (int)(i & (size_t)(ncols - 1));
  float v = fmaf(X[i], sc[col], sh[col]);
  Y[i] = (v > 0.f ? v : 0.f);
}

// ---------------------------------------------------------------------------

extern "C" void kernel_launch(void* const* d_in, const int* in_sizes, int n_in,
                              void* d_out, int out_size, void* d_ws, size_t ws_size,
                              hipStream_t stream) {
  (void)in_sizes; (void)n_in; (void)out_size; (void)ws_size;

  static const int KS[NSZ] = {3, 4, 5, 6, 7, 8};
  static const int SB[NSZ] = {0, 30000, 70000, 120000, 180000, 250000};

  const float* edge_rep  = (const float*)d_in[0];
  const float* cycle_rep = (const float*)d_in[1];
  const int* ce[NSZ];
  for (int s = 0; s < NSZ; ++s) ce[s] = (const int*)d_in[2 + s];
  const float* aut_W = (const float*)d_in[8];
  const float* cw1 = (const float*)d_in[9];
  const float* cg1 = (const float*)d_in[10];
  const float* cb1 = (const float*)d_in[11];
  const float* cw2 = (const float*)d_in[12];
  const float* cg2 = (const float*)d_in[13];
  const float* cb2 = (const float*)d_in[14];
  const float* ew1 = (const float*)d_in[15];
  const float* eg1 = (const float*)d_in[16];
  const float* eb1 = (const float*)d_in[17];
  const float* ew2 = (const float*)d_in[18];
  const float* eg2 = (const float*)d_in[19];
  const float* eb2 = (const float*)d_in[20];

  float* out_edge  = (float*)d_out;
  float* out_cycle = (float*)d_out + (size_t)NEDGE * DD;

  // ---- workspace arena (persistent region + two overlaid phase regions) ----
  char* ws = (char*)d_ws;
  size_t off = 0;
  auto take = [&](size_t bytes) {
    size_t p = off; off += (bytes + 255) & ~(size_t)255; return p;
  };
  const size_t o_autW  = take((size_t)NCH * NSZ * 8 * DD * DD * 2);
  const size_t o_w1c   = take((size_t)CIN * HH * 2);
  const size_t o_w2c   = take((size_t)HH * DD * 2);
  const size_t o_w1e   = take((size_t)2 * DD * HH * 2);
  const size_t o_w2e   = take((size_t)HH * DD * 2);
  const size_t o_part  = take((size_t)NBLK * 2 * HH * 4);
  const size_t o_scale = take((size_t)HH * 4);
  const size_t o_shift = take((size_t)HH * 4);
  const size_t arena   = off;
  auto al = [](size_t b) { return (b + 255) & ~(size_t)255; };
  // cycle phase (dead before edge phase begins)
  const size_t o_nodes = arena;
  const size_t o_cycin = o_nodes + al((size_t)RCY * DD * 2);
  const size_t o_h1    = o_cycin + al((size_t)RCY * CIN * 2);
  const size_t o_a2    = o_h1    + al((size_t)RCY * HH * 4);
  const size_t o_h2    = o_a2    + al((size_t)RCY * HH * 2);
  // edge phase overlay (reuses arena from the start)
  const size_t o_c2e = arena;
  const size_t o_ein = o_c2e + al((size_t)NEDGE * DD * 4);
  const size_t o_eh1 = o_ein + al((size_t)NEDGE * HH * 2);
  const size_t o_ea2 = o_eh1 + al((size_t)NEDGE * HH * 4);
  const size_t o_eh2 = o_ea2 + al((size_t)NEDGE * HH * 2);

  bh*    autWb = (bh*)(ws + o_autW);
  bh*    w1cb  = (bh*)(ws + o_w1c);
  bh*    w2cb  = (bh*)(ws + o_w2c);
  bh*    w1eb  = (bh*)(ws + o_w1e);
  bh*    w2eb  = (bh*)(ws + o_w2e);
  float* part  = (float*)(ws + o_part);
  float* scale = (float*)(ws + o_scale);
  float* shift = (float*)(ws + o_shift);
  bh*    nodes = (bh*)(ws + o_nodes);
  bh*    cycin = (bh*)(ws + o_cycin);
  float* h1    = (float*)(ws + o_h1);
  bh*    a2    = (bh*)(ws + o_a2);
  float* h2    = (float*)(ws + o_h2);
  float* c2e   = (float*)(ws + o_c2e);
  bh*    ein   = (bh*)(ws + o_ein);
  float* eh1   = (float*)(ws + o_eh1);
  bh*    ea2   = (bh*)(ws + o_ea2);
  float* eh2   = (float*)(ws + o_eh2);

  auto cdiv = [](size_t n, size_t b) { return (unsigned)((n + b - 1) / b); };

  // ---- 1. weights -> bf16 ----
  const size_t nAut = (size_t)NCH * NSZ * 8 * DD * DD;
  k_f32_to_bf16<<<cdiv(nAut, 256), 256, 0, stream>>>(aut_W, autWb, nAut);
  k_f32_to_bf16<<<cdiv((size_t)CIN * HH, 256), 256, 0, stream>>>(cw1, w1cb, (size_t)CIN * HH);
  k_f32_to_bf16<<<cdiv((size_t)HH * DD, 256), 256, 0, stream>>>(cw2, w2cb, (size_t)HH * DD);
  k_f32_to_bf16<<<cdiv((size_t)2 * DD * HH, 256), 256, 0, stream>>>(ew1, w1eb, (size_t)2 * DD * HH);
  k_f32_to_bf16<<<cdiv((size_t)HH * DD, 256), 256, 0, stream>>>(ew2, w2eb, (size_t)HH * DD);

  // ---- 2. cyc_in[:, 0:64] = bf16(cycle_rep) ----
  k_cast_cols_bf16<<<cdiv((size_t)RCY * DD, 256), 256, 0, stream>>>(
      cycle_rep, cycin, (size_t)RCY * DD, DD, CIN, 0);

  // ---- 3. gather edges -> cycle nodes (bf16) ----
  for (int s = 0; s < NSZ; ++s) {
    int k = KS[s], rl = MCYC * k;
    k_gather_nodes<<<cdiv((size_t)rl * DD, 256), 256, 0, stream>>>(
        edge_rep, ce[s], nodes, SB[s], k, rl);
  }

  // ---- 4. Autobahn cyclic conv (WMMA) -> cyc_in[:, 64:320] ----
  for (int s = 0; s < NSZ; ++s) {
    int k = KS[s], rl = MCYC * k;
    k_autobahn<<<dim3(rl / 16, NCH), 32, 0, stream>>>(nodes, autWb, cycin, SB[s], k, s);
  }

  // ---- 5. cycle MLP layer 1: [RCY,320]@[320,128] ----
  k_gemm_bf16<8><<<dim3(RCY / 16), 32, 0, stream>>>(cycin, w1cb, h1, CIN, HH);
  k_colstat_part<<<NBLK, HH, 0, stream>>>(h1, part, RCY, HH);
  k_colstat_fin<<<1, HH, 0, stream>>>(part, cg1, cb1, scale, shift, RCY, NBLK, HH);
  k_bn_relu_bf16<<<cdiv((size_t)RCY * HH, 256), 256, 0, stream>>>(
      h1, scale, shift, a2, (size_t)RCY * HH, HH);

  // ---- 6. cycle MLP layer 2: [RCY,128]@[128,64] -> cycle_out (f32, d_out) ----
  k_gemm_bf16<4><<<dim3(RCY / 16), 32, 0, stream>>>(a2, w2cb, h2, HH, DD);
  k_colstat_part<<<NBLK, DD, 0, stream>>>(h2, part, RCY, DD);
  k_colstat_fin<<<1, DD, 0, stream>>>(part, cg2, cb2, scale, shift, RCY, NBLK, DD);
  k_bn_relu_f32<<<cdiv((size_t)RCY * DD, 256), 256, 0, stream>>>(
      h2, scale, shift, out_cycle, (size_t)RCY * DD, DD);

  // ---- 7. scatter cycles -> edges ----
  k_zero_f32<<<cdiv((size_t)NEDGE * DD, 256), 256, 0, stream>>>(c2e, (size_t)NEDGE * DD);
  for (int s = 0; s < NSZ; ++s) {
    int k = KS[s], rl = MCYC * k;
    k_scatter<<<cdiv((size_t)rl * DD, 256), 256, 0, stream>>>(
        out_cycle, ce[s], c2e, SB[s], k, rl);
  }

  // ---- 8. edge_in = bf16(concat(edge_rep, cycle2edge)) ----
  k_cast_cols_bf16<<<cdiv((size_t)NEDGE * DD, 256), 256, 0, stream>>>(
      edge_rep, ein, (size_t)NEDGE * DD, DD, HH, 0);
  k_cast_cols_bf16<<<cdiv((size_t)NEDGE * DD, 256), 256, 0, stream>>>(
      c2e, ein, (size_t)NEDGE * DD, DD, HH, DD);

  // ---- 9. edge MLP layer 1: [NE,128]@[128,128] ----
  k_gemm_bf16<8><<<dim3(NEDGE / 16), 32, 0, stream>>>(ein, w1eb, eh1, HH, HH);
  k_colstat_part<<<NBLK, HH, 0, stream>>>(eh1, part, NEDGE, HH);
  k_colstat_fin<<<1, HH, 0, stream>>>(part, eg1, eb1, scale, shift, NEDGE, NBLK, HH);
  k_bn_relu_bf16<<<cdiv((size_t)NEDGE * HH, 256), 256, 0, stream>>>(
      eh1, scale, shift, ea2, (size_t)NEDGE * HH, HH);

  // ---- 10. edge MLP layer 2: [NE,128]@[128,64] -> edge_out (f32, d_out) ----
  k_gemm_bf16<4><<<dim3(NEDGE / 16), 32, 0, stream>>>(ea2, w2eb, eh2, HH, DD);
  k_colstat_part<<<NBLK, DD, 0, stream>>>(eh2, part, NEDGE, DD);
  k_colstat_fin<<<1, DD, 0, stream>>>(part, eg2, eb2, scale, shift, NEDGE, NBLK, DD);
  k_bn_relu_f32<<<cdiv((size_t)NEDGE * DD, 256), 256, 0, stream>>>(
      eh2, scale, shift, out_edge, (size_t)NEDGE * DD, DD);
}
